// MGLSTM_62680752718329
// MI455X (gfx1250) — matrix operand order repaired and checked by reading, not compiled, not run on hardware
//
#include <hip/hip_runtime.h>
#include <hip/hip_bf16.h>

// ---------------------------------------------------------------------------
// MG-LSTM / AGNN pipeline for MI455X (gfx1250, wave32, WMMA)
//   - xt GEMM and gate GEMM use v_wmma_f32_16x16x32_f16
//   - 9 AGNN propagations factored into: 1 softmax-agg for h, then a single
//     fused 8-channel softmax + 4-gate aggregation pass over edges
// ---------------------------------------------------------------------------

typedef __attribute__((ext_vector_type(16))) _Float16 v16h;
typedef __attribute__((ext_vector_type(8)))  float    v8f;

#define HDIM 96
#define DIN  256

__device__ __forceinline__ float waveReduceSum(float v) {
  // wave32 butterfly reduction
  #pragma unroll
  for (int off = 16; off > 0; off >>= 1) v += __shfl_xor(v, off, 32);
  return v;
}

__device__ __forceinline__ void atomicMaxF32(float* addr, float val) {
  unsigned int* ua = (unsigned int*)addr;
  unsigned int old = __float_as_uint(*((volatile float*)addr));
  while (__uint_as_float(old) < val) {
    unsigned int assumed = old;
    old = atomicCAS(ua, assumed, __float_as_uint(val));
    if (old == assumed) break;
  }
}

// --------------------------------------------------------------------------
// init: fill with -inf-ish for segment max
// --------------------------------------------------------------------------
__global__ void fill_neg(float* __restrict__ p, size_t n) {
  size_t i = (size_t)blockIdx.x * blockDim.x + threadIdx.x;
  if (i < n) p[i] = -3.0e38f;
}

// --------------------------------------------------------------------------
// WMMA GEMM: out[M,96] = A[M,256] @ W[96,256]^T + bias[96]
// One wave per 16x16 output tile; K stepped by 32 (f16 fragments, f32 accum).
// A fragment layout (16-bit A 16x32): lane (g=l>>4,r=l&15): row M=r,
//   VGPR0-3 -> K = g*8 + 0..7, VGPR4-7 -> K = 16 + g*8 + 0..7.
// B fragment (16-bit B 32x16): lane column n=r, same K order as A:
//   elements 0-7 -> K = g*8 + 0..7, elements 8-15 -> K = 16 + g*8 + 0..7.
// --------------------------------------------------------------------------
__global__ void gemm_xt(const float* __restrict__ A, const float* __restrict__ W,
                        const float* __restrict__ bias, float* __restrict__ out,
                        int M) {
  const int lane = threadIdx.x;
  const int r = lane & 15, g = lane >> 4;
  const int m0 = blockIdx.x * 16;
  const int n0 = blockIdx.y * 16;
  int arow = m0 + r; if (arow >= M) arow = M - 1;
  const float* ap = A + (size_t)arow * DIN;
  const float* bp = W + (size_t)(n0 + r) * DIN;

  v8f acc = {};
  #pragma unroll
  for (int kb = 0; kb < DIN; kb += 32) {
    v16h a, b;
    #pragma unroll
    for (int j = 0; j < 8; ++j) {
      a[j]     = (_Float16)ap[kb + g * 8 + j];
      a[j + 8] = (_Float16)ap[kb + 16 + g * 8 + j];
    }
    #pragma unroll
    for (int j = 0; j < 16; ++j)
      b[j] = (_Float16)bp[kb + (j >> 3) * 16 + g * 8 + (j & 7)];
    acc = __builtin_amdgcn_wmma_f32_16x16x32_f16(false, a, false, b,
                                                 (short)0, acc, false, false);
  }
  const float bs = bias[n0 + r];
  // C/D layout: element (vgpr v, lane): M = g*8 + v, N = r
  #pragma unroll
  for (int v = 0; v < 8; ++v) {
    int orow = m0 + g * 8 + v;
    if (orow < M) out[(size_t)orow * HDIM + n0 + r] = acc[v] + bs;
  }
}

// --------------------------------------------------------------------------
// Fused gate GEMM: hN = h + tanh(h@Wb1^T + hN0@Wb2^T + bb1 + bb2)
// K = 96, dual WMMA per K-block sharing one accumulator.
// --------------------------------------------------------------------------
__global__ void gemm_gate(const float* __restrict__ h, const float* __restrict__ hN0,
                          const float* __restrict__ W1, const float* __restrict__ W2,
                          const float* __restrict__ b1, const float* __restrict__ b2,
                          float* __restrict__ hN, int M) {
  const int lane = threadIdx.x;
  const int r = lane & 15, g = lane >> 4;
  const int m0 = blockIdx.x * 16;
  const int n0 = blockIdx.y * 16;
  int arow = m0 + r; if (arow >= M) arow = M - 1;
  const float* a1p = h   + (size_t)arow * HDIM;
  const float* a2p = hN0 + (size_t)arow * HDIM;
  const float* b1p = W1 + (size_t)(n0 + r) * HDIM;
  const float* b2p = W2 + (size_t)(n0 + r) * HDIM;

  v8f acc = {};
  #pragma unroll
  for (int kb = 0; kb < HDIM; kb += 32) {
    v16h a, b;
    #pragma unroll
    for (int j = 0; j < 8; ++j) {
      a[j]     = (_Float16)a1p[kb + g * 8 + j];
      a[j + 8] = (_Float16)a1p[kb + 16 + g * 8 + j];
    }
    #pragma unroll
    for (int j = 0; j < 16; ++j) b[j] = (_Float16)b1p[kb + (j >> 3) * 16 + g * 8 + (j & 7)];
    acc = __builtin_amdgcn_wmma_f32_16x16x32_f16(false, a, false, b,
                                                 (short)0, acc, false, false);
    #pragma unroll
    for (int j = 0; j < 8; ++j) {
      a[j]     = (_Float16)a2p[kb + g * 8 + j];
      a[j + 8] = (_Float16)a2p[kb + 16 + g * 8 + j];
    }
    #pragma unroll
    for (int j = 0; j < 16; ++j) b[j] = (_Float16)b2p[kb + (j >> 3) * 16 + g * 8 + (j & 7)];
    acc = __builtin_amdgcn_wmma_f32_16x16x32_f16(false, a, false, b,
                                                 (short)0, acc, false, false);
  }
  const float bs = b1[n0 + r] + b2[n0 + r];
  #pragma unroll
  for (int v = 0; v < 8; ++v) {
    int orow = m0 + g * 8 + v;
    if (orow < M) {
      size_t idx = (size_t)orow * HDIM + n0 + r;
      hN[idx] = h[idx] + tanhf(acc[v] + bs);
    }
  }
}

// --------------------------------------------------------------------------
// Row L2-normalize (F.normalize semantics): one wave per row, 3 dims/lane
// --------------------------------------------------------------------------
__global__ void rownorm(const float* __restrict__ in, float* __restrict__ out, int N) {
  int row = blockIdx.x * 4 + (threadIdx.x >> 5);
  if (row >= N) return;
  int lane = threadIdx.x & 31;
  const float* p = in + (size_t)row * HDIM;
  float v0 = p[lane], v1 = p[lane + 32], v2 = p[lane + 64];
  float s = waveReduceSum(v0 * v0 + v1 * v1 + v2 * v2);
  float inv = 1.0f / fmaxf(sqrtf(s), 1e-12f);
  float* q = out + (size_t)row * HDIM;
  q[lane] = v0 * inv; q[lane + 32] = v1 * inv; q[lane + 64] = v2 * inv;
}

// edge id helpers: e < E uses edge_index rows; e >= E is self-loop (e-E)
__device__ __forceinline__ void edge_sd(const int* __restrict__ ei, int E, int e,
                                        int& s, int& d) {
  if (e < E) { s = ei[e]; d = ei[E + e]; } else { s = d = e - E; }
}

// --------------------------------------------------------------------------
// Stage-1 edge pass (h): alpha = beta1 * cos(hn[src],hn[dst]); segment max
// --------------------------------------------------------------------------
__global__ void edge_alpha_max(const float* __restrict__ xn, const int* __restrict__ ei,
                               int E, int ET, const float* __restrict__ betas, int bidx,
                               float* __restrict__ alpha, float* __restrict__ amax) {
  int e = blockIdx.x * 4 + (threadIdx.x >> 5);
  if (e >= ET) return;
  int lane = threadIdx.x & 31;
  int s, d; edge_sd(ei, E, e, s, d);
  const float* ps = xn + (size_t)s * HDIM;
  const float* pd = xn + (size_t)d * HDIM;
  float acc = ps[lane] * pd[lane] + ps[lane + 32] * pd[lane + 32] +
              ps[lane + 64] * pd[lane + 64];
  acc = waveReduceSum(acc);
  float a = betas[bidx] * acc;
  if (lane == 0) { alpha[e] = a; atomicMaxF32(&amax[d], a); }
}

__global__ void edge_den(const float* __restrict__ alpha, const int* __restrict__ ei,
                         int E, int ET, const float* __restrict__ amax,
                         float* __restrict__ den) {
  int e = blockIdx.x * 256 + threadIdx.x;
  if (e >= ET) return;
  int d = (e < E) ? ei[E + e] : e - E;
  atomicAdd(&den[d], __expf(alpha[e] - amax[d]));
}

__global__ void edge_agg(const float* __restrict__ feat, const float* __restrict__ alpha,
                         const int* __restrict__ ei, int E, int ET,
                         const float* __restrict__ amax, const float* __restrict__ den,
                         float* __restrict__ out) {
  int e = blockIdx.x * 4 + (threadIdx.x >> 5);
  if (e >= ET) return;
  int lane = threadIdx.x & 31;
  int s, d; edge_sd(ei, E, e, s, d);
  float w = __expf(alpha[e] - amax[d]) / den[d];
  const float* pf = feat + (size_t)s * HDIM;
  float* po = out + (size_t)d * HDIM;
  atomicAdd(&po[lane],      w * pf[lane]);
  atomicAdd(&po[lane + 32], w * pf[lane + 32]);
  atomicAdd(&po[lane + 64], w * pf[lane + 64]);
}

// --------------------------------------------------------------------------
// Stage-2: per-edge cosines for xt and hN (shared by 4 gates each) + 8-channel
// segment max. Channel c <-> betas[2+c]; even c uses cos_xt, odd uses cos_hN.
// --------------------------------------------------------------------------
__global__ void edge_cos2(const float* __restrict__ xtn, const float* __restrict__ hNn,
                          const int* __restrict__ ei, int E, int ET,
                          const float* __restrict__ betas,
                          float* __restrict__ cosx, float* __restrict__ cosh_,
                          float* __restrict__ amax2, int N) {
  int e = blockIdx.x * 4 + (threadIdx.x >> 5);
  if (e >= ET) return;
  int lane = threadIdx.x & 31;
  int s, d; edge_sd(ei, E, e, s, d);
  const float* xs = xtn + (size_t)s * HDIM; const float* xd = xtn + (size_t)d * HDIM;
  const float* hs = hNn + (size_t)s * HDIM; const float* hd = hNn + (size_t)d * HDIM;
  float a1 = xs[lane] * xd[lane] + xs[lane + 32] * xd[lane + 32] +
             xs[lane + 64] * xd[lane + 64];
  float a2 = hs[lane] * hd[lane] + hs[lane + 32] * hd[lane + 32] +
             hs[lane + 64] * hd[lane + 64];
  a1 = waveReduceSum(a1);
  a2 = waveReduceSum(a2);
  if (lane == 0) {
    cosx[e] = a1; cosh_[e] = a2;
    #pragma unroll
    for (int c = 0; c < 8; ++c) {
      float a = betas[2 + c] * ((c & 1) ? a2 : a1);
      atomicMaxF32(&amax2[(size_t)c * N + d], a);
    }
  }
}

__global__ void edge_den2(const float* __restrict__ cosx, const float* __restrict__ cosh_,
                          const int* __restrict__ ei, int E, int ET,
                          const float* __restrict__ betas,
                          const float* __restrict__ amax2, float* __restrict__ den2,
                          int N) {
  int e = blockIdx.x * 256 + threadIdx.x;
  if (e >= ET) return;
  int d = (e < E) ? ei[E + e] : e - E;
  float cx = cosx[e], ch = cosh_[e];
  #pragma unroll
  for (int c = 0; c < 8; ++c) {
    float a = betas[2 + c] * ((c & 1) ? ch : cx);
    atomicAdd(&den2[(size_t)c * N + d], __expf(a - amax2[(size_t)c * N + d]));
  }
}

// Fused 4-gate aggregation: gatepre[g][dst] += w_xt*xt[src] + w_hN*hN[src]
__global__ void edge_agg2(const float* __restrict__ xt, const float* __restrict__ hN,
                          const float* __restrict__ cosx, const float* __restrict__ cosh_,
                          const int* __restrict__ ei, int E, int ET,
                          const float* __restrict__ betas,
                          const float* __restrict__ amax2, const float* __restrict__ den2,
                          float* __restrict__ gp, int N) {
  int e = blockIdx.x * 4 + (threadIdx.x >> 5);
  if (e >= ET) return;
  int lane = threadIdx.x & 31;
  int s, d; edge_sd(ei, E, e, s, d);
  float cx = cosx[e], ch = cosh_[e];
  float w[8];
  #pragma unroll
  for (int c = 0; c < 8; ++c) {
    float a = betas[2 + c] * ((c & 1) ? ch : cx);
    w[c] = __expf(a - amax2[(size_t)c * N + d]) / den2[(size_t)c * N + d];
  }
  const float* xs = xt + (size_t)s * HDIM;
  const float* hs = hN + (size_t)s * HDIM;
  float x0 = xs[lane], x1 = xs[lane + 32], x2 = xs[lane + 64];
  float h0 = hs[lane], h1 = hs[lane + 32], h2 = hs[lane + 64];
  size_t NH = (size_t)N * HDIM;
  #pragma unroll
  for (int gt = 0; gt < 4; ++gt) {
    float wx = w[2 * gt], wh = w[2 * gt + 1];
    float* po = gp + (size_t)gt * NH + (size_t)d * HDIM;
    atomicAdd(&po[lane],      wx * x0 + wh * h0);
    atomicAdd(&po[lane + 32], wx * x1 + wh * h1);
    atomicAdd(&po[lane + 64], wx * x2 + wh * h2);
  }
}

// --------------------------------------------------------------------------
// Finalize: f,i,o = sigmoid(gp0,gp1,gp3); c~ = tanh(gp2)
// c_new = f*c + i*c~; h_new = o*tanh(c_new). out = [h_new | c_new]
// --------------------------------------------------------------------------
__global__ void finalize(const float* __restrict__ gp, const float* __restrict__ c,
                         float* __restrict__ out, int N) {
  size_t tot = (size_t)N * HDIM;
  size_t i = (size_t)blockIdx.x * 256 + threadIdx.x;
  if (i >= tot) return;
  float f  = 1.0f / (1.0f + __expf(-gp[i]));
  float ig = 1.0f / (1.0f + __expf(-gp[tot + i]));
  float ct = tanhf(gp[2 * tot + i]);
  float o  = 1.0f / (1.0f + __expf(-gp[3 * tot + i]));
  float cn = f * c[i] + ig * ct;
  out[i]       = o * tanhf(cn);   // h_new
  out[tot + i] = cn;              // c_new
}

// ---------------------------------------------------------------------------
extern "C" void kernel_launch(void* const* d_in, const int* in_sizes, int n_in,
                              void* d_out, int out_size, void* d_ws, size_t ws_size,
                              hipStream_t stream) {
  const float* x     = (const float*)d_in[0];
  const int*   ei    = (const int*)d_in[1];
  const float* h     = (const float*)d_in[2];
  const float* c     = (const float*)d_in[3];
  const float* W_in  = (const float*)d_in[4];
  const float* b_in  = (const float*)d_in[5];
  const float* Wb1   = (const float*)d_in[10];
  const float* bb1   = (const float*)d_in[11];
  const float* Wb2   = (const float*)d_in[12];
  const float* bb2   = (const float*)d_in[13];
  const float* betas = (const float*)d_in[14];

  const int N  = in_sizes[0] / DIN;
  const int E  = in_sizes[1] / 2;
  const int ET = E + N;
  const size_t NH = (size_t)N * HDIM;

  // workspace layout (floats); tail block is zero-initialized each call
  float* ws    = (float*)d_ws;
  float* xt    = ws;                 // NH
  float* xtn   = xt    + NH;         // NH
  float* hn    = xtn   + NH;         // NH
  float* hN    = hn    + NH;         // NH
  float* hNn   = hN    + NH;         // NH
  float* alpha = hNn   + NH;         // ET
  float* cosx  = alpha + ET;         // ET
  float* coshv = cosx  + ET;         // ET
  float* amaxh = coshv + ET;         // N   (-inf fill)
  float* amax2 = amaxh + N;          // 8N  (-inf fill)
  float* hN0   = amax2 + 8 * (size_t)N;  // NH  (zero)
  float* gp    = hN0   + NH;             // 4NH (zero)
  float* denh  = gp    + 4 * NH;         // N   (zero)
  float* den2  = denh  + N;              // 8N  (zero)

  const size_t zcount = 5 * NH + 9 * (size_t)N;      // hN0..den2
  hipMemsetAsync(hN0, 0, zcount * sizeof(float), stream);
  const size_t fcount = 9 * (size_t)N;               // amaxh..amax2
  fill_neg<<<dim3((unsigned)((fcount + 255) / 256)), 256, 0, stream>>>(amaxh, fcount);

  const dim3 gemmGrid((N + 15) / 16, HDIM / 16);
  const unsigned ewaves = (unsigned)((ET + 3) / 4);
  const unsigned ethreads = (unsigned)((ET + 255) / 256);
  const unsigned nwaves = (unsigned)((N + 3) / 4);

  // Stage 0: xt = x @ W_in^T + b_in  (WMMA)
  gemm_xt<<<gemmGrid, 32, 0, stream>>>(x, W_in, b_in, xt, N);

  // Stage 1: hN0 = agnn(h, beta1)
  rownorm<<<nwaves, 128, 0, stream>>>(h, hn, N);
  edge_alpha_max<<<ewaves, 128, 0, stream>>>(hn, ei, E, ET, betas, 1, alpha, amaxh);
  edge_den<<<ethreads, 256, 0, stream>>>(alpha, ei, E, ET, amaxh, denh);
  edge_agg<<<ewaves, 128, 0, stream>>>(h, alpha, ei, E, ET, amaxh, denh, hN0);

  // hN = h + tanh(h@Wb1^T + hN0@Wb2^T + bb1 + bb2)  (dual WMMA, fused)
  gemm_gate<<<gemmGrid, 32, 0, stream>>>(h, hN0, Wb1, Wb2, bb1, bb2, hN, N);

  // Stage 2: shared cosines + 8 softmax channels + fused 4-gate aggregation
  rownorm<<<nwaves, 128, 0, stream>>>(xt, xtn, N);
  rownorm<<<nwaves, 128, 0, stream>>>(hN, hNn, N);
  edge_cos2<<<ewaves, 128, 0, stream>>>(xtn, hNn, ei, E, ET, betas, cosx, coshv, amax2, N);
  edge_den2<<<ethreads, 256, 0, stream>>>(cosx, coshv, ei, E, ET, betas, amax2, den2, N);
  edge_agg2<<<ewaves, 128, 0, stream>>>(xt, hN, cosx, coshv, ei, E, ET, betas,
                                        amax2, den2, gp, N);

  // Stage 3: elementwise LSTM update -> d_out = [h_new | c_new]
  finalize<<<dim3((unsigned)((NH + 255) / 256)), 256, 0, stream>>>(gp, c, (float*)d_out, N);
}
